// Gemma3Attention_19095424598725
// MI455X (gfx1250) — matrix operand order repaired and checked by reading, not compiled, not run on hardware
//
#include <hip/hip_runtime.h>
#include <hip/hip_bf16.h>

// ---------------- problem constants ----------------
#define BATCH   2
#define SEQ     2048
#define HID     2560
#define NH      8
#define NKV     4
#define HD      256
#define QSZ     (NH * HD)          // 2048
#define KVSZ    (NKV * HD)         // 1024
#define QKVN    (QSZ + 2 * KVSZ)   // 4096
#define SWIN    512
#define SCL     0.0625f            // 256^-0.5
#define MASKVAL (-1.0e30f)

typedef __attribute__((ext_vector_type(16))) __bf16 v16bf;
typedef __attribute__((ext_vector_type(8)))  __bf16 v8bf;
typedef __attribute__((ext_vector_type(8)))  float  v8f;

// CDNA5 async copy: global -> LDS, 16 bytes, tracked by ASYNCcnt.
__device__ __forceinline__ void async_b128(unsigned lds_off, const void* gptr) {
    asm volatile("global_load_async_to_lds_b128 %0, %1, off"
                 :: "v"(lds_off), "v"(gptr) : "memory");
}
__device__ __forceinline__ void wait_async0() {
    asm volatile("s_wait_asynccnt 0x0" ::: "memory");
}

// =====================================================================
// Packing kernels: f32 matrices -> bf16 WMMA-fragment-native layout.
// A-pack:  [M/16][K/32][lane 0..31][e 0..15]   (16x32 A fragments)
//   lane = (k>=16)*16 + (m%16); e -> k = 16*(lane>>4)+e
// B-pack:  [N/16][K/32][lane 0..31][e 0..15]   (32x16 B fragments)
//   lane = (k>=16)*16 + (n%16); e -> k = 16*(lane>>4)+e
// =====================================================================
__global__ __launch_bounds__(256) void pack_a_kernel(
    const float* __restrict__ A, __bf16* __restrict__ Apk, int M, int K)
{
    const int K32 = K >> 5;
    size_t g = (size_t)blockIdx.x * 256 + threadIdx.x;   // one 8-elem group
    int half = (int)(g & 1);
    int lane = (int)((g >> 1) & 31);
    int ks   = (int)((g >> 6) % K32);
    int m16  = (int)((g >> 6) / K32);
    int row  = m16 * 16 + (lane & 15);
    int k0   = ks * 32 + 16 * (lane >> 4) + 8 * half;
    const float* src = A + (size_t)row * K + k0;
    __bf16* dst = Apk + (((size_t)m16 * K32 + ks) * 32 + lane) * 16 + half * 8;
    #pragma unroll
    for (int e = 0; e < 8; ++e) dst[e] = (__bf16)src[e];
}

__global__ __launch_bounds__(256) void pack_b_kernel(
    const float* __restrict__ B, __bf16* __restrict__ Bpk, int K, int N)
{
    const int K32 = K >> 5;
    size_t g = (size_t)blockIdx.x * 256 + threadIdx.x;   // one 16-elem lane
    int lane = (int)(g & 31);
    int ks   = (int)((g >> 5) % K32);
    int n16  = (int)((g >> 5) / K32);
    int n    = n16 * 16 + (lane & 15);
    int kst  = ks * 32 + 16 * (lane >> 4);
    const float* src = B + (size_t)kst * N + n;
    __bf16* dst = Bpk + (((size_t)n16 * K32 + ks) * 32 + lane) * 16;
    #pragma unroll
    for (int e = 0; e < 16; ++e) dst[e] = (__bf16)src[(size_t)e * N];
}

// =====================================================================
// GEMM: C[M,N](f32) = Apk * Bpk, packed-bf16 operands.
// block = 128 thr = 4 waves; tile 64x64; K step 32.
// B tiles (shared by waves) double-buffered in LDS via async copies;
// A fragments (wave-private) loaded straight from global (L2-resident).
// =====================================================================
__global__ __launch_bounds__(128) void gemm_packed_wmma_kernel(
    const __bf16* __restrict__ Apk, const __bf16* __restrict__ Bpk,
    float* __restrict__ C, int M, int N, int K)
{
    __shared__ __align__(32) __bf16 Bt[2][4][32][16];   // 8 KB ping-pong
    const int K32  = K >> 5;
    const int wave = threadIdx.x >> 5;
    const int lane = threadIdx.x & 31;
    const int ntBase = blockIdx.x * 4;                  // 4 col subtiles
    const int rg     = blockIdx.y * 4 + wave;           // wave's 16-row group

    auto issue = [&](int ks, int buf) {                 // this thread copies 32B of subtile j=wave
        const __bf16* src = Bpk + (((size_t)(ntBase + wave) * K32 + ks) * 32 + lane) * 16;
        unsigned dst = (unsigned)(size_t)&Bt[buf][wave][lane][0];
        async_b128(dst,       src);
        async_b128(dst + 16u, src + 8);
    };

    v8f acc[4] = {};
    issue(0, 0);
    for (int ks = 0; ks < K32; ++ks) {
        wait_async0();
        __syncthreads();
        if (ks + 1 < K32) issue(ks + 1, (ks + 1) & 1);
        const int buf = ks & 1;
        v16bf a = *(const v16bf*)(Apk + (((size_t)rg * K32 + ks) * 32 + lane) * 16);
        #pragma unroll
        for (int j = 0; j < 4; ++j) {
            v16bf b = *(const v16bf*)&Bt[buf][j][lane][0];
            acc[j] = __builtin_amdgcn_wmma_f32_16x16x32_bf16(
                false, a, false, b, (short)0, acc[j], false, false);
        }
    }

    // C layout: lane col = lane&15, VGPR r -> row r + 8*(lane>=16)
    const int m  = lane & 15;
    const int mh = (lane >> 4) * 8;
    #pragma unroll
    for (int j = 0; j < 4; ++j)
        #pragma unroll
        for (int r = 0; r < 8; ++r)
            C[(size_t)(rg * 16 + mh + r) * N + ((ntBase + j) * 16 + m)] = acc[j][r];
}

// =====================================================================
// RMSNorm + RoPE; writes Q/K/V directly in WMMA-fragment-packed bf16.
// Q: A-frag pack per (b,h):  [S/16][8][32][16]
// K: B-frag pack per (b,hk): [S/16][8][32][16]   (n = key, k = d)
// V: B-frag pack per (b,hk): [S/32][16][32][16]  (n = d,   k = key)
// =====================================================================
__device__ __forceinline__ size_t q_pack_off(int s, int d) {
    int m16 = s >> 4, mm = s & 15;
    int dc = d >> 5, k = d & 31;
    int lh = k >> 4;
    int e  = k & 15;
    int ln = lh * 16 + mm;
    return ((size_t)(m16 * 8 + dc) * 32 + ln) * 16 + e;
}
__device__ __forceinline__ size_t k_pack_off(int s, int d) {
    int m16 = s >> 4, n = s & 15;
    int dc = d >> 5, k = d & 31;
    int ln = (k >> 4) * 16 + n;
    return ((size_t)(m16 * 8 + dc) * 32 + ln) * 16 + (k & 15);
}
__device__ __forceinline__ size_t v_pack_off(int s, int d) {
    int kc = s >> 5, lh = (s >> 4) & 1, e = s & 15;
    int n0 = d >> 4, n = d & 15;
    int ln = lh * 16 + n;
    return ((size_t)(kc * 16 + n0) * 32 + ln) * 16 + e;
}

__global__ __launch_bounds__(256) void norm_rope_split_kernel(
    const float* __restrict__ qkv,
    const float* __restrict__ qw, const float* __restrict__ kw,
    const int*   __restrict__ pos,
    __bf16* __restrict__ Qpk, __bf16* __restrict__ Kpk, __bf16* __restrict__ Vpk)
{
    const int bs = blockIdx.x;
    const int b  = bs / SEQ, s = bs % SEQ;
    const int d  = threadIdx.x;
    const int warp = d >> 5;

    __shared__ float red[8];
    __shared__ float vecs[HD];

    const float* row = qkv + (size_t)bs * QKVN;
    const float p   = (float)pos[s];
    const float ang = p * __powf(10000.f, -(float)(2 * (d & 127)) / 256.f);
    const float cs  = __cosf(ang);
    const float sn  = __sinf(ang);

    const size_t headQ = (size_t)(SEQ / 16) * 8 * 512;   // elems per Q head
    const size_t headK = headQ;
    const size_t headV = (size_t)(SEQ / 32) * 16 * 512;

    // ---- Q heads ----
    for (int h = 0; h < NH; ++h) {
        float x  = row[h * HD + d];
        float ss = x * x;
        #pragma unroll
        for (int off = 16; off > 0; off >>= 1) ss += __shfl_xor(ss, off, 32);
        if ((d & 31) == 0) red[warp] = ss;
        __syncthreads();
        float tot = 0.f;
        #pragma unroll
        for (int i = 0; i < 8; ++i) tot += red[i];
        float rms = rsqrtf(tot / (float)HD + 1e-6f);
        float nx  = x * rms * (1.f + qw[d]);
        vecs[d] = nx;
        __syncthreads();
        float partner = (d < 128) ? -vecs[d + 128] : vecs[d - 128];
        Qpk[(size_t)(b * NH + h) * headQ + q_pack_off(s, d)] = (__bf16)(nx * cs + partner * sn);
        __syncthreads();
    }
    // ---- K heads ----
    for (int h = 0; h < NKV; ++h) {
        float x  = row[QSZ + h * HD + d];
        float ss = x * x;
        #pragma unroll
        for (int off = 16; off > 0; off >>= 1) ss += __shfl_xor(ss, off, 32);
        if ((d & 31) == 0) red[warp] = ss;
        __syncthreads();
        float tot = 0.f;
        #pragma unroll
        for (int i = 0; i < 8; ++i) tot += red[i];
        float rms = rsqrtf(tot / (float)HD + 1e-6f);
        float nx  = x * rms * (1.f + kw[d]);
        vecs[d] = nx;
        __syncthreads();
        float partner = (d < 128) ? -vecs[d + 128] : vecs[d - 128];
        Kpk[(size_t)(b * NKV + h) * headK + k_pack_off(s, d)] = (__bf16)(nx * cs + partner * sn);
        __syncthreads();
    }
    // ---- V heads ----
    for (int h = 0; h < NKV; ++h)
        Vpk[(size_t)(b * NKV + h) * headV + v_pack_off(s, d)]
            = (__bf16)row[QSZ + KVSZ + h * HD + d];
}

// =====================================================================
// Sliding-window flash attention. grid=(B*H, S/64); 128 thr = 4 waves.
// K/V tiles (32 KB each, shared across waves) async-staged into LDS;
// all WMMA fragments are contiguous b128-pair loads.
// =====================================================================
__global__ __launch_bounds__(128) void attn_swin_kernel(
    const __bf16* __restrict__ Qpk, const __bf16* __restrict__ Kpk,
    const __bf16* __restrict__ Vpk, float* __restrict__ Ob)
{
    const int bh = blockIdx.x;
    const int b  = bh >> 3;
    const int h  = bh & 7;
    const int hk = h >> 1;
    const int qt = blockIdx.y;

    const int wave = threadIdx.x >> 5;
    const int lane = threadIdx.x & 31;
    const int m    = lane & 15;
    const int kb   = (lane >> 4) * 8;
    const int mh   = (lane >> 4) * 8;

    const int tile_q0 = qt * 64;
    const int q0 = tile_q0 + wave * 16;
    const int rg = q0 >> 4;                 // wave's 16-row fragment group

    const __bf16* Qp = Qpk + (size_t)(b * NH  + h ) * (SEQ / 16) * 8 * 512;
    const __bf16* Kp = Kpk + (size_t)(b * NKV + hk) * (SEQ / 16) * 8 * 512;
    const __bf16* Vp = Vpk + (size_t)(b * NKV + hk) * (SEQ / 32) * 16 * 512;

    __shared__ __align__(32) __bf16 Kt[4 * 8 * 512];    // 32 KB: 64-key K tile
    __shared__ __align__(32) __bf16 Vt[2 * 16 * 512];   // 32 KB: 64-key V tile
    __shared__ __align__(32) __bf16 Plds[4][16][64];    // 8 KB: per-wave P staging
    const unsigned KtB = (unsigned)(size_t)&Kt[0];
    const unsigned VtB = (unsigned)(size_t)&Vt[0];

    v8f o[16] = {};
    float mrow[8], lrow[8];
    #pragma unroll
    for (int r = 0; r < 8; ++r) { mrow[r] = MASKVAL; lrow[r] = 0.f; }

    int k_lo = tile_q0 - (SWIN - 1);
    if (k_lo < 0) k_lo = 0;
    k_lo &= ~63;
    const int k_hi = tile_q0 + 63;

    for (int ks = k_lo; ks <= k_hi; ks += 64) {
        // ---- async-stage K/V tiles (contiguous in packed layout) ----
        {
            const __bf16* ksrc = Kp + (size_t)(ks >> 4) * 8 * 512;
            const __bf16* vsrc = Vp + (size_t)(ks >> 5) * 16 * 512;
            for (int i = threadIdx.x; i < 2048; i += 128) {   // 2048 x 16B = 32KB
                async_b128(KtB + i * 16, ksrc + (size_t)i * 8);
                async_b128(VtB + i * 16, vsrc + (size_t)i * 8);
            }
        }
        wait_async0();
        __syncthreads();

        // ---- S = Q * K^T (16 rows x 64 keys, D = 256) ----
        v8f sacc[4] = {};
        for (int dc = 0; dc < 8; ++dc) {
            v16bf qa = *(const v16bf*)(Qp + ((size_t)(rg * 8 + dc) * 32 + lane) * 16);
            #pragma unroll
            for (int j = 0; j < 4; ++j) {
                v16bf kf = *(const v16bf*)&Kt[((j * 8 + dc) * 32 + lane) * 16];
                sacc[j] = __builtin_amdgcn_wmma_f32_16x16x32_bf16(
                    false, qa, false, kf, (short)0, sacc[j], false, false);
            }
        }

        // ---- mask + scale + tile row-max ----
        float tmax[8];
        #pragma unroll
        for (int r = 0; r < 8; ++r) tmax[r] = MASKVAL;
        #pragma unroll
        for (int j = 0; j < 4; ++j) {
            int key = ks + j * 16 + m;
            #pragma unroll
            for (int r = 0; r < 8; ++r) {
                int qi = q0 + mh + r;
                float v = sacc[j][r] * SCL;
                bool ok = (key <= qi) && (qi - key < SWIN);
                v = ok ? v : MASKVAL;
                sacc[j][r] = v;
                tmax[r] = fmaxf(tmax[r], v);
            }
        }
        #pragma unroll
        for (int r = 0; r < 8; ++r) {
            float t = tmax[r];
            t = fmaxf(t, __shfl_xor(t, 1, 32));
            t = fmaxf(t, __shfl_xor(t, 2, 32));
            t = fmaxf(t, __shfl_xor(t, 4, 32));
            t = fmaxf(t, __shfl_xor(t, 8, 32));
            tmax[r] = t;
        }

        float alpha[8], psum[8];
        #pragma unroll
        for (int r = 0; r < 8; ++r) {
            float mnew = fmaxf(mrow[r], tmax[r]);
            alpha[r] = __expf(mrow[r] - mnew);
            mrow[r]  = mnew;
            psum[r]  = 0.f;
        }

        // ---- P = exp(S - m); stage to per-wave LDS (C->A transpose) ----
        #pragma unroll
        for (int j = 0; j < 4; ++j)
            #pragma unroll
            for (int r = 0; r < 8; ++r) {
                float pv = __expf(sacc[j][r] - mrow[r]);
                psum[r] += pv;
                Plds[wave][mh + r][j * 16 + m] = (__bf16)pv;
            }

        #pragma unroll
        for (int r = 0; r < 8; ++r) {
            float t = psum[r];
            t += __shfl_xor(t, 1, 32);
            t += __shfl_xor(t, 2, 32);
            t += __shfl_xor(t, 4, 32);
            t += __shfl_xor(t, 8, 32);
            lrow[r] = lrow[r] * alpha[r] + t;
        }

        #pragma unroll
        for (int n0 = 0; n0 < 16; ++n0)
            #pragma unroll
            for (int r = 0; r < 8; ++r)
                o[n0][r] *= alpha[r];

        // ---- O += P * V (K-dim = 64 keys in 2 chunks of 32) ----
        #pragma unroll
        for (int kc = 0; kc < 2; ++kc) {
            v8bf p0 = *(const v8bf*)&Plds[wave][m][kc * 32 + 2 * kb];
            v8bf p1 = *(const v8bf*)&Plds[wave][m][kc * 32 + 2 * kb + 8];
            v16bf pa = __builtin_shufflevector(p0, p1,
                0, 1, 2, 3, 4, 5, 6, 7, 8, 9, 10, 11, 12, 13, 14, 15);
            #pragma unroll
            for (int n0 = 0; n0 < 16; ++n0) {
                v16bf vf = *(const v16bf*)&Vt[((kc * 16 + n0) * 32 + lane) * 16];
                o[n0] = __builtin_amdgcn_wmma_f32_16x16x32_bf16(
                    false, pa, false, vf, (short)0, o[n0], false, false);
            }
        }
        __syncthreads();   // all waves done reading before next tile overwrites
    }

    // ---- epilogue: normalize, store f32 [B, S, H*D] ----
    #pragma unroll
    for (int n0 = 0; n0 < 16; ++n0)
        #pragma unroll
        for (int r = 0; r < 8; ++r) {
            int qi = q0 + mh + r;
            Ob[((size_t)b * SEQ + qi) * QSZ + h * HD + n0 * 16 + m]
                = o[n0][r] / lrow[r];
        }
}

// =====================================================================
// host launcher
// =====================================================================
extern "C" void kernel_launch(void* const* d_in, const int* in_sizes, int n_in,
                              void* d_out, int out_size, void* d_ws, size_t ws_size,
                              hipStream_t stream) {
    const float* hidden = (const float*)d_in[0];   // [B,S,HID]
    const float* w_qkv  = (const float*)d_in[1];   // [HID, QKVN]
    const float* w_o    = (const float*)d_in[2];   // [QSZ, HID]
    const float* qnw    = (const float*)d_in[3];   // [HD]
    const float* knw    = (const float*)d_in[4];   // [HD]
    const int*   pos    = (const int*)d_in[5];     // [S]
    float* out = (float*)d_out;                    // [B,S,HID]

    const int M = BATCH * SEQ;                     // 4096

    // workspace layout with region reuse (total 136 MB)
    char* ws = (char*)d_ws;
    const size_t MB = 1024ull * 1024ull;
    float*  qkv  = (float*)(ws + 0);               // [0,64) MB; dead after norm
    __bf16* Apk1 = (__bf16*)(ws + 64 * MB);        // [64,84); dead after gemm1
    __bf16* Bpk1 = (__bf16*)(ws + 84 * MB);        // [84,104); dead after gemm1
    __bf16* Qpk  = (__bf16*)(ws + 104 * MB);       // [104,120)
    __bf16* Kpk  = (__bf16*)(ws + 120 * MB);       // [120,128)
    __bf16* Vpk  = (__bf16*)(ws + 128 * MB);       // [128,136)
    float*  att  = (float*)(ws + 64 * MB);         // reuse [64,96)
    __bf16* Apk2 = (__bf16*)(ws + 0);              // reuse [0,16)
    __bf16* Bpk2 = (__bf16*)(ws + 16 * MB);        // reuse [16,26)

    // 1) pack hidden (A) and w_qkv (B)
    pack_a_kernel<<<(size_t)M * HID / 8 / 256, 256, 0, stream>>>(hidden, Apk1, M, HID);
    pack_b_kernel<<<(size_t)HID * QKVN / 16 / 256, 256, 0, stream>>>(w_qkv, Bpk1, HID, QKVN);

    // 2) QKV projection: [4096,2560] x [2560,4096] -> f32
    {
        dim3 grid(QKVN / 64, M / 64), block(128);
        gemm_packed_wmma_kernel<<<grid, block, 0, stream>>>(Apk1, Bpk1, qkv, M, QKVN, HID);
    }
    // 3) RMSNorm + RoPE -> packed bf16 Q/K/V
    norm_rope_split_kernel<<<M, HD, 0, stream>>>(qkv, qnw, knw, pos, Qpk, Kpk, Vpk);

    // 4) sliding-window flash attention -> f32 [B,S,QSZ]
    {
        dim3 grid(BATCH * NH, SEQ / 64), block(128);
        attn_swin_kernel<<<grid, block, 0, stream>>>(Qpk, Kpk, Vpk, att);
    }
    // 5) pack attention output (A) and w_o (B); reuse freed regions
    pack_a_kernel<<<(size_t)M * QSZ / 8 / 256, 256, 0, stream>>>(att, Apk2, M, QSZ);
    pack_b_kernel<<<(size_t)QSZ * HID / 16 / 256, 256, 0, stream>>>(w_o, Bpk2, QSZ, HID);

    // 6) output projection: [4096,2048] x [2048,2560] -> d_out
    {
        dim3 grid(HID / 64, M / 64), block(128);
        gemm_packed_wmma_kernel<<<grid, block, 0, stream>>>(Apk2, Bpk2, out, M, HID, QSZ);
    }
}
